// RAE_24455543783986
// MI455X (gfx1250) — matrix operand order repaired and checked by reading, not compiled, not run on hardware
//
#include <hip/hip_runtime.h>
#include <math.h>

// ---------------- problem constants ----------------
#define DD        512
#define TWO_D     1024
#define DEPTH     18
#define N_NODES   ((1 << DEPTH) - 1)            // 262143
#define N_INTERNAL ((1 << (DEPTH - 1)) - 1)     // 131071
#define N_LEAVES  (1 << (DEPTH - 1))            // 131072
#define M_TILE    32

// ---------------- types ----------------
typedef __attribute__((ext_vector_type(16))) __bf16 v16bf;
typedef __attribute__((ext_vector_type(8)))  float  v8f;
typedef __attribute__((ext_vector_type(4)))  unsigned int u32x4;
typedef __attribute__((ext_vector_type(8)))  unsigned int u32x8;

static __device__ __forceinline__ unsigned short f32_to_bf16(float f) {
    unsigned int u = __builtin_bit_cast(unsigned int, f);
    u += 0x7FFFu + ((u >> 16) & 1u);   // round to nearest even
    return (unsigned short)(u >> 16);
}

struct Pair128 { uint4 a, b; };
static __device__ __forceinline__ v16bf make_frag(uint4 lo, uint4 hi) {
    Pair128 p; p.a = lo; p.b = hi;
    return __builtin_bit_cast(v16bf, p);
}

// Branch-free tanh: (e^2x - 1)/(e^2x + 1), clamped; v_exp_f32 + v_rcp_f32.
static __device__ __forceinline__ float fast_tanh(float x) {
    float xc = fminf(fmaxf(x, -9.0f), 9.0f);
    float t  = __expf(2.0f * xc);
    return (t - 1.0f) * __builtin_amdgcn_rcpf(t + 1.0f);
}

// ---------------------------------------------------------------------------
// Kernel 1: pack W (1024x512 f32, K-major) into bf16 B-fragments.
// Layout: [kb 0..31][nb 0..31][lane 0..31][e 0..15] contiguous bf16.
// ---------------------------------------------------------------------------
__global__ void __launch_bounds__(256) pack_w_kernel(
    const float* __restrict__ W, unsigned short* __restrict__ Wp) {
    int tid = blockIdx.x * blockDim.x + threadIdx.x;     // 32768 groups
    if (tid >= 32 * 32 * 32) return;
    int lane = tid & 31;
    int nbg  = (tid >> 5) & 31;
    int kb   = tid >> 10;
    int col   = nbg * 16 + (lane & 15);
    int kbase = kb * 32 + ((lane & 16) ? 16 : 0);
    unsigned short* dst = Wp + (size_t)tid * 16;
#pragma unroll
    for (int e = 0; e < 16; ++e)
        dst[e] = f32_to_bf16(W[(size_t)(kbase + e) * DD + col]);
}

// ---------------------------------------------------------------------------
// Kernel 2: convert leaf rows of inp (f32) to bf16 h-buffer.
// ---------------------------------------------------------------------------
__global__ void __launch_bounds__(256) cvt_leaves_kernel(
    const float* __restrict__ inp, unsigned short* __restrict__ h) {
    size_t i = ((size_t)blockIdx.x * blockDim.x + threadIdx.x) * 4;
    const size_t total = (size_t)N_LEAVES * DD;
    if (i >= total) return;
    const size_t base = (size_t)N_INTERNAL * DD;
    float4 f = *(const float4*)(inp + base + i);
    unsigned int p0 = (unsigned int)f32_to_bf16(f.x) |
                      ((unsigned int)f32_to_bf16(f.y) << 16);
    unsigned int p1 = (unsigned int)f32_to_bf16(f.z) |
                      ((unsigned int)f32_to_bf16(f.w) << 16);
    uint2 pk; pk.x = p0; pk.y = p1;
    *(uint2*)(h + base + i) = pk;
}

// ---------------------------------------------------------------------------
// Kernel 3: one tree level.  Block = 256 threads (8 wave32); tile = 32 rows x
// 512 cols.  A-tile (32 rows x 1024) staged into LDS by the Tensor Data Mover
// in GATHER mode (8 descriptors x 8 row indices).  LDS: [side][row][512 bf16].
// Wave w owns n-blocks [4w,4w+4) across both 16-row sub-tiles: each B fragment
// feeds 2 WMMAs.  B fragments are double-buffered one k-block ahead so the
// loads of block kb+1 are in flight during block kb's 8 WMMAs.
// ---------------------------------------------------------------------------
__global__ void __launch_bounds__(256) level_gemm_kernel(
    unsigned short* __restrict__ h,            // bf16 node states (read+write)
    const unsigned short* __restrict__ Wp,     // packed bf16 W fragments
    const int* __restrict__ left,
    const int* __restrict__ right,
    const float* __restrict__ bias,
    int lvl_start, int m_lvl,
    float* __restrict__ out)                   // non-null only for level 0
{
    __shared__ unsigned short As[2 * M_TILE * DD];  // 64 KB

    const int tid  = threadIdx.x;
    const int wave = tid >> 5;
    const int lane = tid & 31;
    const int row_base = blockIdx.x * M_TILE;
    const int nb0  = wave * 4;                // first of 4 n-blocks (16 cols)

    // B fragments viewed as whole 32-byte objects (32B-aligned): the compiler
    // lowers each to two aligned global_load_b128 against an SGPR base.
    const v16bf* Bp = (const v16bf*)Wp;

    // Warm L2/L0 with this wave's first B-fragment region (global_prefetch_b8).
    __builtin_prefetch(Bp + ((unsigned)nb0 * 32u + (unsigned)lane), 0, 1);

    // ---- TDM gather: stage A tile (zero VGPR traffic), wave 0 only ----
    if (wave == 0) {
        unsigned int lds_base =
            __builtin_amdgcn_readfirstlane((unsigned int)(unsigned long long)(uintptr_t)As);
        unsigned long long ga = (unsigned long long)(uintptr_t)h;
        unsigned int ga_lo = __builtin_amdgcn_readfirstlane((unsigned int)ga);
        unsigned int ga_hi = __builtin_amdgcn_readfirstlane((unsigned int)(ga >> 32));
#pragma unroll
        for (int side = 0; side < 2; ++side) {
            const int* childArr = side ? right : left;   // side0 = cols[0,512) = left
#pragma unroll
            for (int d = 0; d < 4; ++d) {
                unsigned int idx[8];
#pragma unroll
                for (int i = 0; i < 8; ++i) {
                    int row  = row_base + d * 8 + i;
                    int rowc = (row < m_lvl) ? row : (m_lvl - 1);
                    idx[i] = __builtin_amdgcn_readfirstlane(
                        (unsigned int)childArr[lvl_start + rowc]);
                }
                // D# group0: count=1 | gather_index_size=32b | gather_mode
                u32x4 g0;
                g0[0] = 0xC0000001u;
                g0[1] = lds_base + (unsigned)side * 32768u + (unsigned)d * 8192u;
                g0[2] = ga_lo;
                g0[3] = (ga_hi & 0x01FFFFFFu) | (2u << 30);   // type=2 (image)
                // D# group1: data_size=2B; tensor_dim0=512; tensor_dim1=N_NODES;
                // tile_dim0=512; tile_dim1=8 indices; dim0_stride=512.
                u32x8 g1;
                g1[0] = 0x00010000u;            // data_size=1 (2 bytes)
                g1[1] = 0x02000000u;            // tensor_dim0 low16 = 512
                g1[2] = 0xFFFF0000u;            // tensor_dim1 low16 (N_NODES)
                g1[3] = 0x02000003u;            // tensor_dim1 hi=3 | tile_dim0=512
                g1[4] = 0x00000008u;            // tile_dim1 = 8 valid indices
                g1[5] = 512u;                   // tensor_dim0_stride low32
                g1[6] = 0u;
                g1[7] = 0u;
                u32x4 g2; g2[0]=idx[0]; g2[1]=idx[1]; g2[2]=idx[2]; g2[3]=idx[3];
                u32x4 g3; g3[0]=idx[4]; g3[1]=idx[5]; g3[2]=idx[6]; g3[3]=idx[7];
                asm volatile("tensor_load_to_lds %0, %1, %2, %3"
                             :: "s"(g0), "s"(g1), "s"(g2), "s"(g3)
                             : "memory");
            }
        }
        __builtin_amdgcn_s_wait_tensorcnt(0);
    }
    __syncthreads();

    // A fragment addressing (contiguous k order, matching packed B):
    // lanes 0-15: row=lane,  k [0..15]
    // lanes16-31: row=lane-16, k [16..31]
    const int arow  = lane & 15;
    const int abase = (lane & 16) ? 16 : 0;

    v8f acc[2][4] = {{v8f{}, v8f{}, v8f{}, v8f{}},
                     {v8f{}, v8f{}, v8f{}, v8f{}}};

    auto loadB = [&](int kb, v16bf bf[4]) {
        unsigned base = ((unsigned)kb * 32u + (unsigned)nb0) * 32u + (unsigned)lane;
#pragma unroll
        for (int j = 0; j < 4; ++j)
            bf[j] = Bp[base + (unsigned)j * 32u];
    };
    auto loadA = [&](int kb, v16bf af[2]) {
        const unsigned short* sbase = As + (kb >> 4) * (M_TILE * DD);
        const int kk = (kb & 15) * 32;
#pragma unroll
        for (int rt = 0; rt < 2; ++rt) {
            const unsigned short* ap = sbase + (rt * 16 + arow) * DD + kk;
            af[rt] = make_frag(*(const uint4*)(ap + abase),
                               *(const uint4*)(ap + abase + 8));
        }
    };
    auto domma = [&](const v16bf af[2], const v16bf bf[4]) {
#pragma unroll
        for (int j = 0; j < 4; ++j) {
#pragma unroll
            for (int rt = 0; rt < 2; ++rt) {
                acc[rt][j] = __builtin_amdgcn_wmma_f32_16x16x32_bf16(
                    false, af[rt], false, bf[j], (short)0, acc[rt][j],
                    false, false);
            }
        }
    };

    v16bf b0[4], b1[4], af[2];
    loadB(0, b0);
    for (int kb = 0; kb < 32; kb += 2) {
        loadB(kb + 1, b1);                       // in flight during b0's WMMAs
        loadA(kb, af);
        domma(af, b0);
        loadB(kb + 2 < 32 ? kb + 2 : 0, b0);     // in flight during b1's WMMAs
        loadA(kb + 1, af);
        domma(af, b1);
    }

    // ---- epilogue: bias + fast tanh, store bf16 per C layout ----
    // C layout: element v of v8f -> M = v + (lane>=16 ? 8 : 0), N = lane&15.
    const int col   = lane & 15;
    const int rbase = (lane & 16) ? 8 : 0;
#pragma unroll
    for (int rt = 0; rt < 2; ++rt) {
#pragma unroll
        for (int j = 0; j < 4; ++j) {
            int ncol = (nb0 + j) * 16 + col;
            float bv = bias[ncol];
#pragma unroll
            for (int v = 0; v < 8; ++v) {
                int r   = rt * 16 + rbase + v;
                int row = row_base + r;
                if (row < m_lvl) {
                    float val = fast_tanh(acc[rt][j][v] + bv);
                    int node  = lvl_start + row;
                    h[(size_t)node * DD + ncol] = f32_to_bf16(val);
                    if (out && row == 0) out[ncol] = val;   // root == node 0
                }
            }
        }
    }
}

// ---------------------------------------------------------------------------
extern "C" void kernel_launch(void* const* d_in, const int* in_sizes, int n_in,
                              void* d_out, int out_size, void* d_ws, size_t ws_size,
                              hipStream_t stream) {
    (void)in_sizes; (void)n_in; (void)out_size; (void)ws_size;
    const int*   left  = (const int*)d_in[0];
    const int*   right = (const int*)d_in[1];
    // d_in[2] = is_leaf (unused), d_in[4] = root (== 0)
    const float* inp   = (const float*)d_in[3];
    const float* W     = (const float*)d_in[5];
    const float* b     = (const float*)d_in[6];
    float*       out   = (float*)d_out;

    // workspace: [0, 1MB) packed bf16 W ; [1MB, ...) bf16 h for all nodes
    unsigned short* Wp = (unsigned short*)d_ws;                 // 524288 bf16
    unsigned short* h  = (unsigned short*)d_ws + (1 << 19);     // N_NODES*512 bf16

    pack_w_kernel<<<(32 * 32 * 32 + 255) / 256, 256, 0, stream>>>(W, Wp);

    {
        size_t total = (size_t)N_LEAVES * DD / 4;               // 16M threads
        cvt_leaves_kernel<<<(unsigned)((total + 255) / 256), 256, 0, stream>>>(inp, h);
    }

    for (int lvl = DEPTH - 2; lvl >= 0; --lvl) {
        int m     = 1 << lvl;
        int start = (1 << lvl) - 1;
        int blocks = (m + M_TILE - 1) / M_TILE;
        level_gemm_kernel<<<blocks, 256, 0, stream>>>(
            h, Wp, left, right, b, start, m, (lvl == 0) ? out : nullptr);
    }
}
